// Token_GAT_24979529794139
// MI455X (gfx1250) — matrix operand order repaired and checked by reading, not compiled, not run on hardware
//
#include <hip/hip_runtime.h>
#include <hip/hip_bf16.h>

typedef __attribute__((ext_vector_type(16))) _Float16 v16h;
typedef __attribute__((ext_vector_type(8)))  float    v8f;

#define BATCH 32
#define NN    512
#define DIN   768
#define DOUT  256
#define HEADS 8
#define PSTRIDE 516   // 512 + 4 floats pad -> distinct LDS banks per row

static __device__ __forceinline__ v8f wmma16(v16h a, v16h b, v8f c) {
  return __builtin_amdgcn_wmma_f32_16x16x32_f16(false, a, false, b,
                                                (short)0, c, false, false);
}

// ---------------------------------------------------------------------------
// Fragment layout (CDNA5 ISA 7.12.2, wave32):
//  A 16x32 f16: lane carries row m = lane&15; element e -> VGPR v=e/2, half h=e&1
//               k = (v<4 ? 0:16) + (lane>=16 ? 8:0) + 2*(v&3) + h
//  B 32x16 f16: lane carries col n = lane&15; k = (e<8 ? 0:16) + (lane>=16 ? 8:0) + (e&7)
//  C/D 16x16 f32: acc[r] holds m = r + (lane>=16 ? 8:0), n = lane&15
// ---------------------------------------------------------------------------

__global__ __launch_bounds__(256) void zero_f32(float* p, int n) {
  int i = blockIdx.x * 256 + threadIdx.x;
  if (i < n) p[i] = 0.0f;
}

// Pack row-major f32 [M x K] into A-fragment f16 tiles: [Mt][Kt][lane][16]
__global__ __launch_bounds__(256) void pack_a(const float* __restrict__ src,
                                              _Float16* __restrict__ dst, int K) {
  int Kt = K >> 5;
  size_t gid = (size_t)blockIdx.x * 256 + threadIdx.x;
  int tile   = (int)(gid >> 9);
  int within = (int)(gid & 511);
  int lane = within >> 4;
  int e    = within & 15;
  int mt = tile / Kt, kt = tile - mt * Kt;
  int m = mt * 16 + (lane & 15);
  int v = e >> 1, hh = e & 1;
  int k = kt * 32 + ((v < 4) ? 0 : 16) + ((lane >= 16) ? 8 : 0) + ((v & 3) << 1) + hh;
  dst[gid] = (_Float16)src[(size_t)m * K + k];
}

// Pack batched row-major f32 [batches][K x N] into B-fragment tiles:
// [batch][Kt][Nt][lane][16]
__global__ __launch_bounds__(256) void pack_b(const float* __restrict__ src,
                                              _Float16* __restrict__ dst,
                                              int K, int Ncols) {
  int Kt = K >> 5, Nt = Ncols >> 4;
  size_t perBatch = (size_t)Kt * Nt * 512;
  size_t gid = (size_t)blockIdx.x * 256 + threadIdx.x;
  size_t bidx = gid / perBatch;
  size_t rem  = gid - bidx * perBatch;
  int tile   = (int)(rem >> 9);
  int within = (int)(rem & 511);
  int lane = within >> 4;
  int e    = within & 15;
  int kt = tile / Nt, nt = tile - kt * Nt;
  int n = nt * 16 + (lane & 15);
  int k = kt * 32 + ((e < 8) ? 0 : 16) + ((lane >= 16) ? 8 : 0) + (e & 7);
  dst[gid] = (_Float16)src[bidx * (size_t)K * Ncols + (size_t)k * Ncols + n];
}

// C[Mt*16 x Nt*16] (f32, ldc = Nt*16) = Apack x Bpack.
// Register-blocked 2x2: block = 8 waves covers 32 rows x 256 cols;
// wave owns 2 M-tiles x 2 N-tiles -> 4 WMMAs per 4 fragment loads.
// Requires Ntiles == 16; grid.y = Mtiles/2.
__global__ __launch_bounds__(256) void gemm_wmma(const _Float16* __restrict__ Apack,
                                                 const _Float16* __restrict__ Bpack,
                                                 float* __restrict__ C,
                                                 int Ktiles, int Ntiles) {
  int t = threadIdx.x, lane = t & 31, wave = t >> 5;
  int mt0 = blockIdx.y * 2;
  int nt0 = wave * 2;
  const _Float16* a0p = Apack + ((size_t)mt0 * Ktiles) * 512 + (size_t)lane * 16;
  const _Float16* a1p = a0p + (size_t)Ktiles * 512;
  const _Float16* bp  = Bpack + (size_t)lane * 16;
  v8f acc00 = {}, acc01 = {}, acc10 = {}, acc11 = {};
  for (int kt = 0; kt < Ktiles; ++kt) {
    v16h a0 = *(const v16h*)(a0p + (size_t)kt * 512);
    v16h a1 = *(const v16h*)(a1p + (size_t)kt * 512);
    v16h b0 = *(const v16h*)(bp + ((size_t)kt * Ntiles + nt0) * 512);
    v16h b1 = *(const v16h*)(bp + ((size_t)kt * Ntiles + nt0 + 1) * 512);
    if (kt + 1 < Ktiles)
      __builtin_prefetch(bp + (((size_t)kt + 1) * Ntiles + nt0) * 512, 0, 0);
    acc00 = wmma16(a0, b0, acc00);
    acc01 = wmma16(a0, b1, acc01);
    acc10 = wmma16(a1, b0, acc10);
    acc11 = wmma16(a1, b1, acc11);
  }
  int ldc  = Ntiles * 16;
  int mrow = ((lane >= 16) ? 8 : 0);
  int c0   = nt0 * 16 + (lane & 15);
#pragma unroll
  for (int r = 0; r < 8; ++r) {
    size_t row0 = (size_t)(mt0 * 16 + mrow + r) * ldc;
    size_t row1 = (size_t)((mt0 + 1) * 16 + mrow + r) * ldc;
    C[row0 + c0]      = acc00[r];
    C[row0 + c0 + 16] = acc01[r];
    C[row1 + c0]      = acc10[r];
    C[row1 + c0 + 16] = acc11[r];
  }
}

// f1[row] = dot(Wh[row,:], a1); f2[row] = dot(Wh[row,:], a2). One block per row.
__global__ __launch_bounds__(256) void scores_kernel(const float* __restrict__ Wh,
                                                     const float* __restrict__ a1,
                                                     const float* __restrict__ a2,
                                                     float* __restrict__ f1,
                                                     float* __restrict__ f2) {
  __shared__ float s1[256], s2[256];
  int row = blockIdx.x, t = threadIdx.x;
  float w = Wh[(size_t)row * DOUT + t];
  s1[t] = w * a1[t];
  s2[t] = w * a2[t];
  __syncthreads();
  for (int off = 128; off > 0; off >>= 1) {
    if (t < off) { s1[t] += s1[t + off]; s2[t] += s2[t + off]; }
    __syncthreads();
  }
  if (t == 0) { f1[row] = s1[0]; f2[row] = s2[0]; }
}

// Masked row-softmax attention + aggregation for one (batch, 16-row strip).
// Normalization (1/rowsum) is folded into the epilogue, so the exp pass emits
// f16 probabilities directly into LDS in A-fragment order; the WMMA loop then
// reads A via contiguous ds_load_b128 and feeds two accumulators per fragment.
// mode 0: out += elu(attn@Wh) * scale   (multi-head mean accumulation)
// mode 1: out  = relu(attn@Wh)          (final layer)
__global__ __launch_bounds__(256) void attn_kernel(const float* __restrict__ f1,
                                                   const float* __restrict__ f2,
                                                   const int* __restrict__ adj,
                                                   const _Float16* __restrict__ WhBpack,
                                                   float* __restrict__ out,
                                                   int mode, float scale) {
  __shared__ float scoresLDS[16 * PSTRIDE];              // 33 KB
  __shared__ __align__(16) _Float16 afrag[16 * 512];     // 16 KB, [kt][lane][e]
  __shared__ float red[16 * 16];
  __shared__ float rowmax[16];
  __shared__ float rowinv[16];

  int b = blockIdx.y;
  int row0 = blockIdx.x * 16;
  int t = threadIdx.x;
  int r = t >> 4;       // row within strip (16 threads per row)
  int c16 = t & 15;

  // ---- pass 1: e = leaky_relu(f1_i + f2_j), mask, row max ----
  float fi = f1[b * NN + row0 + r];
  const int* adjRow = adj + ((size_t)(b * NN + row0 + r)) * NN;
  float lmax = -3.0e38f;
  for (int i = 0; i < 32; ++i) {
    int j = c16 + i * 16;
    float s = fi + f2[b * NN + j];
    s = s > 0.0f ? s : 0.2f * s;
    if (adjRow[j] == 0) s = -9.0e15f;
    scoresLDS[r * PSTRIDE + j] = s;
    lmax = fmaxf(lmax, s);
  }
  red[r * 16 + c16] = lmax;
  __syncthreads();
  if (t < 16) {
    float m = red[t * 16];
    for (int i = 1; i < 16; ++i) m = fmaxf(m, red[t * 16 + i]);
    rowmax[t] = m;
  }
  __syncthreads();

  // ---- pass 2: exp; write f16 directly in A-fragment order; row sum ----
  float rm = rowmax[r];
  float lsum = 0.0f;
  for (int i = 0; i < 32; ++i) {
    int j = c16 + i * 16;
    float ev = __expf(scoresLDS[r * PSTRIDE + j] - rm);
    lsum += ev;
    // inverse of A-fragment mapping: j -> (kt, lane, e)
    int kt     = j >> 5;
    int k32    = j & 31;
    int hh     = k32 & 1;
    int laneHi = (k32 >> 3) & 1;
    int v      = (((k32 >> 4) & 1) << 2) | ((k32 >> 1) & 3);
    int lane_  = (laneHi << 4) | r;
    int e      = (v << 1) | hh;
    afrag[(kt << 9) + (lane_ << 4) + e] = (_Float16)ev;
  }
  red[r * 16 + c16] = lsum;
  __syncthreads();
  if (t < 16) {
    float s = 0.0f;
    for (int i = 0; i < 16; ++i) s += red[t * 16 + i];
    rowinv[t] = 1.0f / s;
  }
  __syncthreads();

  // ---- WMMA: (16x512 unnormalized probs) x (512x256 Wh), dual accumulators ----
  int lane = t & 31, wave = t >> 5;
  int nt0 = wave, nt1 = wave + 8;
  const _Float16* af = afrag + (size_t)lane * 16;
  const _Float16* bp = WhBpack + (((size_t)b * 16) * 16) * 512 + (size_t)lane * 16;
  v8f acc0 = {}, acc1 = {};
  for (int kt = 0; kt < 16; ++kt) {
    v16h a  = *(const v16h*)(af + (size_t)kt * 512);
    v16h b0 = *(const v16h*)(bp + ((size_t)kt * 16 + nt0) * 512);
    v16h b1 = *(const v16h*)(bp + ((size_t)kt * 16 + nt1) * 512);
    acc0 = wmma16(a, b0, acc0);
    acc1 = wmma16(a, b1, acc1);
  }

  // ---- epilogue: fold softmax normalization, activation, store ----
  int mbase = ((lane >= 16) ? 8 : 0);
#pragma unroll
  for (int rr = 0; rr < 8; ++rr) {
    int m = rr + mbase;
    float invm = rowinv[m];
    size_t base = ((size_t)b * NN + row0 + m) * DOUT + (lane & 15);
    float v0 = acc0[rr] * invm;
    float v1 = acc1[rr] * invm;
    if (mode == 0) {
      v0 = v0 > 0.0f ? v0 : (__expf(v0) - 1.0f);   // elu
      v1 = v1 > 0.0f ? v1 : (__expf(v1) - 1.0f);
      out[base + nt0 * 16] += v0 * scale;
      out[base + nt1 * 16] += v1 * scale;
    } else {
      out[base + nt0 * 16] = v0 > 0.0f ? v0 : 0.0f;  // relu
      out[base + nt1 * 16] = v1 > 0.0f ? v1 : 0.0f;
    }
  }
}

extern "C" void kernel_launch(void* const* d_in, const int* in_sizes, int n_in,
                              void* d_out, int out_size, void* d_ws, size_t ws_size,
                              hipStream_t stream) {
  const float* x    = (const float*)d_in[0];   // [B,N,Din]
  const int*   adj  = (const int*)d_in[1];     // [B,N,N]
  const float* W0   = (const float*)d_in[2];   // [H,Din,Dout]
  const float* a1_0 = (const float*)d_in[3];   // [H,Dout]
  const float* a2_0 = (const float*)d_in[4];   // [H,Dout]
  const float* Wo   = (const float*)d_in[5];   // [Dout,Dout]
  const float* a1_o = (const float*)d_in[6];   // [Dout]
  const float* a2_o = (const float*)d_in[7];   // [Dout]
  float* out = (float*)d_out;                  // [B,N,Dout]

  const size_t M = (size_t)BATCH * NN;         // 16384 rows

  // ---- workspace carve-up (all 32B aligned) ----
  char* ws = (char*)d_ws;
  _Float16* Apack_x = (_Float16*)ws;                         // M*DIN halves  (24 MB)
  _Float16* BpackW0 = Apack_x + M * DIN;                     // H*DIN*DOUT    (3 MB)
  float*    Wh      = (float*)(BpackW0 + (size_t)HEADS * DIN * DOUT); // M*DOUT f32 (16 MB)
  float*    f1      = Wh + M * DOUT;                         // M f32
  float*    f2      = f1 + M;                                // M f32
  _Float16* WhBpack = (_Float16*)(f2 + M);                   // B*NN*DOUT halves (8 MB)
  float*    h_mean  = (float*)(WhBpack + (size_t)BATCH * NN * DOUT); // M*DOUT f32 (16 MB)
  _Float16* Apack_h = (_Float16*)(h_mean + M * DOUT);        // M*DOUT halves (8 MB)
  _Float16* BpackWo = Apack_h + M * DOUT;                    // DOUT*DOUT halves

  // ---- one-time packs ----
  {
    int n = (int)(M * DOUT);
    zero_f32<<<dim3((n + 255) / 256), dim3(256), 0, stream>>>(h_mean, n);
  }
  pack_a<<<dim3((unsigned)(M * DIN / 256)), dim3(256), 0, stream>>>(x, Apack_x, DIN);
  pack_b<<<dim3((unsigned)((size_t)HEADS * DIN * DOUT / 256)), dim3(256), 0, stream>>>(
      W0, BpackW0, DIN, DOUT);

  // ---- layer 1: 8 heads, mean(elu(.)) accumulated into h_mean ----
  for (int h = 0; h < HEADS; ++h) {
    gemm_wmma<<<dim3(1, (unsigned)(M / 32)), dim3(256), 0, stream>>>(
        Apack_x, BpackW0 + (size_t)h * DIN * DOUT, Wh, DIN / 32, DOUT / 16);
    scores_kernel<<<dim3((unsigned)M), dim3(256), 0, stream>>>(
        Wh, a1_0 + h * DOUT, a2_0 + h * DOUT, f1, f2);
    pack_b<<<dim3((unsigned)((size_t)BATCH * NN * DOUT / 256)), dim3(256), 0, stream>>>(
        Wh, WhBpack, NN, DOUT);   // batched: 32 matrices of [512 x 256]
    attn_kernel<<<dim3(NN / 16, BATCH), dim3(256), 0, stream>>>(
        f1, f2, adj, WhBpack, h_mean, 0, 1.0f / HEADS);
  }

  // ---- layer 2 ----
  pack_a<<<dim3((unsigned)(M * DOUT / 256)), dim3(256), 0, stream>>>(h_mean, Apack_h, DOUT);
  pack_b<<<dim3((unsigned)((size_t)DOUT * DOUT / 256)), dim3(256), 0, stream>>>(
      Wo, BpackWo, DOUT, DOUT);
  gemm_wmma<<<dim3(1, (unsigned)(M / 32)), dim3(256), 0, stream>>>(
      Apack_h, BpackWo, Wh, DOUT / 32, DOUT / 16);
  scores_kernel<<<dim3((unsigned)M), dim3(256), 0, stream>>>(Wh, a1_o, a2_o, f1, f2);
  pack_b<<<dim3((unsigned)((size_t)BATCH * NN * DOUT / 256)), dim3(256), 0, stream>>>(
      Wh, WhBpack, NN, DOUT);
  attn_kernel<<<dim3(NN / 16, BATCH), dim3(256), 0, stream>>>(
      f1, f2, adj, WhBpack, out, 1, 1.0f);
}
